// Sketch_RNN_60464549593816
// MI455X (gfx1250) — matrix operand order repaired and checked by reading, not compiled, not run on hardware
//
#include <hip/hip_runtime.h>
#include <hip/hip_bf16.h>

#define SEQL  128
#define BATCH 512
#define PTD   5
#define EHD   256
#define ZSD   128
#define DHD   512
#define NMIX  20
#define NMAXT 256

typedef __attribute__((ext_vector_type(16))) _Float16 v16h;
typedef __attribute__((ext_vector_type(8)))  _Float16 h8;
typedef __attribute__((ext_vector_type(8)))  float    v8f;

#define WMMA_F16(A, Bm, C) \
  __builtin_amdgcn_wmma_f32_16x16x32_f16(false, (A), false, (Bm), (short)0, (C), false, false)

__device__ __forceinline__ float sigf(float x) { return 1.0f / (1.0f + __expf(-x)); }

// ---- A fragment: 16x32 f16, rows m0..m0+15 (batch), cols k0..k0+31 from row-major X[ld=ldx]
// ISA layout: lanes 0-15 -> M=lane, K = 0..7 (v0-3) and 16..23 (v4-7);
//             lanes 16-31 -> M=lane-16, K = 8..15 and 24..31.
__device__ __forceinline__ v16h load_A(const _Float16* __restrict__ X, int ldx,
                                       int m0, int k0, int lane) {
  int m  = m0 + (lane & 15);
  int kb = k0 + ((lane >> 4) << 3);
  const _Float16* p = X + (size_t)m * ldx + kb;
  v16h a;
  *(h8*)&a       = *(const h8*)p;         // K offsets +0..7
  *((h8*)&a + 1) = *(const h8*)(p + 16);  // K offsets +16..23
  return a;
}

// ---- B fragment: 32x16 f16 from pre-packed PB[((kc*Npad + n)*2 + half)*16 + j]
// ISA layout: lanes 0-15 -> col n, K=0..7 and 16..23; lanes 16-31 -> col n, K=8..15 and 24..31.
__device__ __forceinline__ v16h load_B(const _Float16* __restrict__ PB, int Npad,
                                       int kc, int n0, int lane) {
  int n    = n0 + (lane & 15);
  int half = lane >> 4;
  const _Float16* p = PB + ((((size_t)kc * Npad + n) * 2 + half) << 4);
  v16h b;
  *(h8*)&b       = *(const h8*)p;
  *((h8*)&b + 1) = *(const h8*)(p + 8);
  return b;
}

// ---- Pack weight W[Nreal][Kreal] (f32 row-major) into WMMA-B fragment order (f16, zero-padded)
__global__ void pack_w_kernel(const float* __restrict__ W, _Float16* __restrict__ PB,
                              int Nreal, int Kreal, int Npad, int KcTot) {
  int idx = blockIdx.x * blockDim.x + threadIdx.x;
  int total = KcTot * Npad * 32;
  if (idx >= total) return;
  int j    = idx & 15;
  int half = (idx >> 4) & 1;
  int n    = (idx >> 5) % Npad;
  int kc   = (idx >> 5) / Npad;
  int k    = kc * 32 + ((j & 8) << 1) + half * 8 + (j & 7);
  float v  = (n < Nreal && k < Kreal) ? W[(size_t)n * Kreal + k] : 0.0f;
  PB[idx] = (_Float16)v;
}

// ---- Pack encoder inputs s[SEQ][B][5] -> xpad[SEQ][B][32] f16 (zero padded K)
__global__ void pack_xenc_kernel(const float* __restrict__ s, _Float16* __restrict__ xp) {
  int idx = blockIdx.x * blockDim.x + threadIdx.x;
  int total = SEQL * BATCH * 32;
  if (idx >= total) return;
  int kk = idx & 31;
  int tb = idx >> 5;             // t*BATCH + b
  float v = (kk < PTD) ? s[(size_t)tb * PTD + kk] : 0.0f;
  xp[idx] = (_Float16)v;
}

__global__ void zero_f16_kernel(_Float16* p, int n) {
  int i = blockIdx.x * blockDim.x + threadIdx.x;
  if (i < n) p[i] = (_Float16)0.0f;
}
__global__ void zero_f32_kernel(float* p, int n) {
  int i = blockIdx.x * blockDim.x + threadIdx.x;
  if (i < n) p[i] = 0.0f;
}

// xpad_dec[b][0..4] = s0 = {0,0,1,0,0}; tail [133..159] = 0 (z slots 5..132 filled later)
__global__ void init_xdec_kernel(_Float16* __restrict__ xp) {
  int idx = blockIdx.x * blockDim.x + threadIdx.x;
  if (idx >= BATCH * 160) return;
  int k = idx % 160;
  if (k < PTD)       xp[idx] = (_Float16)((k == 2) ? 1.0f : 0.0f);
  else if (k >= 133) xp[idx] = (_Float16)0.0f;
}

// ================= Encoder: fused 4-gate GEMM (WMMA) + LSTM cell, both directions ========
// Each wave: M=32 (two row tiles) x N=16 hidden cols, all 4 gates -> 8 accumulators.
// B fragments are reused by 2 WMMAs each (halved weight traffic, better latency hiding).
__launch_bounds__(128)
__global__ void enc_step_kernel(const _Float16* __restrict__ xpad,
                                const _Float16* __restrict__ pih_f, const _Float16* __restrict__ phh_f,
                                const float* __restrict__ bias_f,
                                const _Float16* __restrict__ pih_b, const _Float16* __restrict__ phh_b,
                                const float* __restrict__ bias_b,
                                const _Float16* __restrict__ hin_f, _Float16* __restrict__ hout_f,
                                float* __restrict__ c_f,
                                const _Float16* __restrict__ hin_b, _Float16* __restrict__ hout_b,
                                float* __restrict__ c_b,
                                int t) {
  const int dir  = blockIdx.y;
  const int wave = threadIdx.x >> 5;
  const int lane = threadIdx.x & 31;
  const int tile = blockIdx.x * 4 + wave;          // 256 tiles = 16 (batch/32) x 16 (hidden/16)
  const int nt = tile & 15, mt = tile >> 4;
  const int m0 = mt * 32, n0 = nt * 16;

  const _Float16* pih  = dir ? pih_b  : pih_f;
  const _Float16* phh  = dir ? phh_b  : phh_f;
  const float*    bias = dir ? bias_b : bias_f;
  const _Float16* hin  = dir ? hin_b  : hin_f;
  _Float16*       hout = dir ? hout_b : hout_f;
  float*          cbuf = dir ? c_b    : c_f;
  const int tx = dir ? (SEQL - 1 - t) : t;
  const _Float16* X = xpad + (size_t)tx * BATCH * 32;

  v8f vzero = {0.f, 0.f, 0.f, 0.f, 0.f, 0.f, 0.f, 0.f};
  v8f acc[4][2];
#pragma unroll
  for (int g = 0; g < 4; ++g) { acc[g][0] = vzero; acc[g][1] = vzero; }

  // input projection: one K=32 chunk (PT=5 zero-padded)
  {
    v16h a0 = load_A(X, 32, m0, 0, lane);
    v16h a1 = load_A(X, 32, m0 + 16, 0, lane);
#pragma unroll
    for (int g = 0; g < 4; ++g) {
      v16h bm = load_B(pih, 4 * EHD, 0, g * EHD + n0, lane);
      acc[g][0] = WMMA_F16(a0, bm, acc[g][0]);
      acc[g][1] = WMMA_F16(a1, bm, acc[g][1]);
    }
  }
  // recurrent projection: K = 256 -> 8 chunks
#pragma unroll
  for (int kc = 0; kc < 8; ++kc) {
    v16h a0 = load_A(hin, EHD, m0, kc * 32, lane);
    v16h a1 = load_A(hin, EHD, m0 + 16, kc * 32, lane);
#pragma unroll
    for (int g = 0; g < 4; ++g) {
      v16h bm = load_B(phh, 4 * EHD, kc, g * EHD + n0, lane);
      acc[g][0] = WMMA_F16(a0, bm, acc[g][0]);
      acc[g][1] = WMMA_F16(a1, bm, acc[g][1]);
    }
  }

  // LSTM cell (gate order i, f, g, o)
  const int n = n0 + (lane & 15);
  const float bi = bias[n], bf = bias[EHD + n], bg = bias[2 * EHD + n], bo = bias[3 * EHD + n];
  const int mbase = m0 + ((lane >> 4) << 3);
#pragma unroll
  for (int rt = 0; rt < 2; ++rt) {
#pragma unroll
    for (int r = 0; r < 8; ++r) {
      int m = mbase + rt * 16 + r;
      size_t off = (size_t)m * EHD + n;
      float ig = sigf(acc[0][rt][r] + bi);
      float fg = sigf(acc[1][rt][r] + bf);
      float gg = tanhf(acc[2][rt][r] + bg);
      float og = sigf(acc[3][rt][r] + bo);
      float c = fg * cbuf[off] + ig * gg;
      cbuf[off] = c;
      hout[off] = (_Float16)(og * tanhf(c));
    }
  }
}

// ===== Latent: mu/sigma/z (small, VALU); also fills z into decoder input pad (slots 5..132)
__global__ void latent_kernel(const _Float16* __restrict__ hf, const _Float16* __restrict__ hb,
                              const float* __restrict__ Wmu, const float* __restrict__ bmu,
                              const float* __restrict__ Wsig, const float* __restrict__ bsig,
                              const float* __restrict__ z_eps,
                              float* __restrict__ zbuf, _Float16* __restrict__ xpd) {
  int idx = blockIdx.x * blockDim.x + threadIdx.x;
  if (idx >= BATCH * ZSD) return;
  int j = idx % ZSD, b = idx / ZSD;
  float am = bmu[j], as = bsig[j];
  const float* wm = Wmu  + (size_t)j * (2 * EHD);
  const float* ws = Wsig + (size_t)j * (2 * EHD);
  for (int k = 0; k < EHD; ++k) {
    float h = (float)hf[(size_t)b * EHD + k];
    am += h * wm[k];  as += h * ws[k];
  }
  for (int k = 0; k < EHD; ++k) {
    float h = (float)hb[(size_t)b * EHD + k];
    am += h * wm[EHD + k];  as += h * ws[EHD + k];
  }
  float z = am + __expf(0.5f * as) * z_eps[idx];
  zbuf[idx] = z;
  xpd[(size_t)b * 160 + PTD + j] = (_Float16)z;
}

// ===== h0 = tanh(z @ Wh0.T + bh0) -> decoder hidden (f16)
__global__ void h0_kernel(const float* __restrict__ zbuf, const float* __restrict__ Wh0,
                          const float* __restrict__ bh0, _Float16* __restrict__ h0) {
  int idx = blockIdx.x * blockDim.x + threadIdx.x;
  if (idx >= BATCH * DHD) return;
  int j = idx % DHD, b = idx / DHD;
  float a = bh0[j];
  const float* w = Wh0 + (size_t)j * ZSD;
  const float* z = zbuf + (size_t)b * ZSD;
  for (int k = 0; k < ZSD; ++k) a += z[k] * w[k];
  h0[idx] = (_Float16)tanhf(a);
}

// ================= Decoder: fused 4-gate GEMM (WMMA) + LSTM cell, M=32 per wave ==========
__launch_bounds__(128)
__global__ void dec_step_kernel(const _Float16* __restrict__ xpd,
                                const _Float16* __restrict__ pih, const _Float16* __restrict__ phh,
                                const float* __restrict__ bias,
                                const _Float16* __restrict__ hin, _Float16* __restrict__ hout,
                                float* __restrict__ cbuf) {
  const int wave = threadIdx.x >> 5;
  const int lane = threadIdx.x & 31;
  const int tile = blockIdx.x * 4 + wave;           // 512 tiles = 16 (batch/32) x 32 (hidden/16)
  const int nt = tile & 31, mt = tile >> 5;
  const int m0 = mt * 32, n0 = nt * 16;

  v8f vzero = {0.f, 0.f, 0.f, 0.f, 0.f, 0.f, 0.f, 0.f};
  v8f acc[4][2];
#pragma unroll
  for (int g = 0; g < 4; ++g) { acc[g][0] = vzero; acc[g][1] = vzero; }

  // input projection: x = [prev(5), z(128)] zero-padded to K=160 -> 5 chunks
#pragma unroll
  for (int kc = 0; kc < 5; ++kc) {
    v16h a0 = load_A(xpd, 160, m0, kc * 32, lane);
    v16h a1 = load_A(xpd, 160, m0 + 16, kc * 32, lane);
#pragma unroll
    for (int g = 0; g < 4; ++g) {
      v16h bm = load_B(pih, 4 * DHD, kc, g * DHD + n0, lane);
      acc[g][0] = WMMA_F16(a0, bm, acc[g][0]);
      acc[g][1] = WMMA_F16(a1, bm, acc[g][1]);
    }
  }
  // recurrent projection: K = 512 -> 16 chunks
#pragma unroll 4
  for (int kc = 0; kc < 16; ++kc) {
    if (kc < 15)  // stream next B-fragment line into cache (global_prefetch_b8)
      __builtin_prefetch((const void*)(phh + (((size_t)(kc + 1) * (4 * DHD) + n0) << 5)), 0, 1);
    v16h a0 = load_A(hin, DHD, m0, kc * 32, lane);
    v16h a1 = load_A(hin, DHD, m0 + 16, kc * 32, lane);
#pragma unroll
    for (int g = 0; g < 4; ++g) {
      v16h bm = load_B(phh, 4 * DHD, kc, g * DHD + n0, lane);
      acc[g][0] = WMMA_F16(a0, bm, acc[g][0]);
      acc[g][1] = WMMA_F16(a1, bm, acc[g][1]);
    }
  }

  const int n = n0 + (lane & 15);
  const float bi = bias[n], bf = bias[DHD + n], bg = bias[2 * DHD + n], bo = bias[3 * DHD + n];
  const int mbase = m0 + ((lane >> 4) << 3);
#pragma unroll
  for (int rt = 0; rt < 2; ++rt) {
#pragma unroll
    for (int r = 0; r < 8; ++r) {
      int m = mbase + rt * 16 + r;
      size_t off = (size_t)m * DHD + n;
      float ig = sigf(acc[0][rt][r] + bi);
      float fg = sigf(acc[1][rt][r] + bf);
      float gg = tanhf(acc[2][rt][r] + bg);
      float og = sigf(acc[3][rt][r] + bo);
      float c = fg * cbuf[off] + ig * gg;
      cbuf[off] = c;
      hout[off] = (_Float16)(og * tanhf(c));
    }
  }
}

// ===== Decoder head: y = h@Wdec.T + b via WMMA into LDS, then GMM sampling per row =======
// Block covers 32 batch rows x 128 (padded) cols; wave w owns 16 cols, 2 row-tiles.
__launch_bounds__(256)
__global__ void dec_out_kernel(const _Float16* __restrict__ h,
                               const _Float16* __restrict__ pwd, const float* __restrict__ bdec,
                               const float* __restrict__ gmm_eps, const float* __restrict__ pen_u,
                               float* __restrict__ out, _Float16* __restrict__ xpd, int t) {
  __shared__ float ytile[32][132];
  const int wave = threadIdx.x >> 5;
  const int lane = threadIdx.x & 31;
  const int m0 = blockIdx.x * 32;   // 16 blocks cover B=512
  const int n0 = wave * 16;         // 8 waves cover 128 cols (123 real)

  v8f vzero = {0.f, 0.f, 0.f, 0.f, 0.f, 0.f, 0.f, 0.f};
  v8f acc0 = vzero, acc1 = vzero;
#pragma unroll 4
  for (int kc = 0; kc < 16; ++kc) {
    v16h a0 = load_A(h, DHD, m0, kc * 32, lane);
    v16h a1 = load_A(h, DHD, m0 + 16, kc * 32, lane);
    v16h bm = load_B(pwd, 128, kc, n0, lane);
    acc0 = WMMA_F16(a0, bm, acc0);
    acc1 = WMMA_F16(a1, bm, acc1);
  }
  const int n = n0 + (lane & 15);
  const float bb = (n < 123) ? bdec[n] : 0.0f;
  const int rbase = (lane >> 4) << 3;
#pragma unroll
  for (int r = 0; r < 8; ++r) {
    ytile[rbase + r][n]      = acc0[r] + bb;
    ytile[rbase + r + 16][n] = acc1[r] + bb;
  }
  __syncthreads();

  if (threadIdx.x < 32) {
    const int row = threadIdx.x;
    const int b = m0 + row;
    const float* y = ytile[row];
    // pi = softmax(y[m*6]) over NM mixtures
    float pmax = -3.0e38f;
    for (int m = 0; m < NMIX; ++m) pmax = fmaxf(pmax, y[m * 6]);
    float pw[NMIX], psum = 0.0f;
    for (int m = 0; m < NMIX; ++m) { pw[m] = __expf(y[m * 6] - pmax); psum += pw[m]; }
    float inv = 1.0f / psum;
    const float* eps = gmm_eps + (((size_t)t * BATCH + b) * NMIX) * 2;
    float dx = 0.0f, dy = 0.0f;
    for (int m = 0; m < NMIX; ++m) {
      float pi  = pw[m] * inv;
      float mx  = y[m * 6 + 1], my = y[m * 6 + 2];
      float sx  = __expf(y[m * 6 + 3]), sy = __expf(y[m * 6 + 4]);
      float cor = tanhf(y[m * 6 + 5]);
      float l21 = cor / sx;
      float l22 = sqrtf(fmaxf(sy * sy - l21 * l21, 1e-6f));
      float e1 = eps[m * 2], e2 = eps[m * 2 + 1];
      dx += pi * (mx + sx * e1);
      dy += pi * (my + l21 * e1 + l22 * e2);
    }
    // pen = (u < softmax(y[120..122]))
    float q0 = y[120], q1 = y[121], q2 = y[122];
    float qm = fmaxf(q0, fmaxf(q1, q2));
    float e0 = __expf(q0 - qm), e1 = __expf(q1 - qm), e2 = __expf(q2 - qm);
    float qinv = 1.0f / (e0 + e1 + e2);
    const float* u = pen_u + ((size_t)t * BATCH + b) * 3;
    float p0 = (u[0] < e0 * qinv) ? 1.0f : 0.0f;
    float p1 = (u[1] < e1 * qinv) ? 1.0f : 0.0f;
    float p2 = (u[2] < e2 * qinv) ? 1.0f : 0.0f;
    float* o = out + ((size_t)t * BATCH + b) * 5;
    o[0] = dx; o[1] = dy; o[2] = p0; o[3] = p1; o[4] = p2;
    _Float16* xp = xpd + (size_t)b * 160;
    xp[0] = (_Float16)dx; xp[1] = (_Float16)dy;
    xp[2] = (_Float16)p0; xp[3] = (_Float16)p1; xp[4] = (_Float16)p2;
  }
}

extern "C" void kernel_launch(void* const* d_in, const int* in_sizes, int n_in,
                              void* d_out, int out_size, void* d_ws, size_t ws_size,
                              hipStream_t stream) {
  (void)in_sizes; (void)n_in; (void)out_size;
  const float* s        = (const float*)d_in[0];
  const float* encWih_f = (const float*)d_in[1];
  const float* encWhh_f = (const float*)d_in[2];
  const float* encb_f   = (const float*)d_in[3];
  const float* encWih_b = (const float*)d_in[4];
  const float* encWhh_b = (const float*)d_in[5];
  const float* encb_b   = (const float*)d_in[6];
  const float* Wmu      = (const float*)d_in[7];
  const float* bmu      = (const float*)d_in[8];
  const float* Wsig     = (const float*)d_in[9];
  const float* bsig     = (const float*)d_in[10];
  const float* Wh0      = (const float*)d_in[11];
  const float* bh0      = (const float*)d_in[12];
  const float* decWih   = (const float*)d_in[13];
  const float* decWhh   = (const float*)d_in[14];
  const float* decb     = (const float*)d_in[15];
  const float* Wdec     = (const float*)d_in[16];
  const float* bdec     = (const float*)d_in[17];
  const float* z_eps    = (const float*)d_in[18];
  const float* gmm_eps  = (const float*)d_in[19];
  const float* pen_u    = (const float*)d_in[20];
  float* out = (float*)d_out;

  char* ws = (char*)d_ws;
  size_t off = 0;
  auto take = [&](size_t bytes) -> char* {
    off = (off + 255) & ~(size_t)255;
    char* p = ws + off;
    off += bytes;
    return p;
  };
  _Float16* xpad_enc  = (_Float16*)take((size_t)SEQL * BATCH * 32 * 2);
  _Float16* p_eih_f   = (_Float16*)take((size_t)1  * 1024 * 32 * 2);
  _Float16* p_eih_b   = (_Float16*)take((size_t)1  * 1024 * 32 * 2);
  _Float16* p_ehh_f   = (_Float16*)take((size_t)8  * 1024 * 32 * 2);
  _Float16* p_ehh_b   = (_Float16*)take((size_t)8  * 1024 * 32 * 2);
  _Float16* henc_f0   = (_Float16*)take((size_t)BATCH * EHD * 2);
  _Float16* henc_f1   = (_Float16*)take((size_t)BATCH * EHD * 2);
  _Float16* henc_b0   = (_Float16*)take((size_t)BATCH * EHD * 2);
  _Float16* henc_b1   = (_Float16*)take((size_t)BATCH * EHD * 2);
  float*    cenc      = (float*)   take((size_t)2 * BATCH * EHD * 4);
  _Float16* p_dih     = (_Float16*)take((size_t)5  * 2048 * 32 * 2);
  _Float16* p_dhh     = (_Float16*)take((size_t)16 * 2048 * 32 * 2);
  _Float16* p_wdec    = (_Float16*)take((size_t)16 * 128  * 32 * 2);
  _Float16* xpad_dec  = (_Float16*)take((size_t)BATCH * 160 * 2);
  _Float16* hdec0     = (_Float16*)take((size_t)BATCH * DHD * 2);
  _Float16* hdec1     = (_Float16*)take((size_t)BATCH * DHD * 2);
  float*    cdec      = (float*)   take((size_t)BATCH * DHD * 4);
  float*    zbuf      = (float*)   take((size_t)BATCH * ZSD * 4);
  if (off > ws_size) return;

  float* cenc_f = cenc;
  float* cenc_b = cenc + (size_t)BATCH * EHD;

  auto cdiv = [](long nn, long dd) { return (unsigned)((nn + dd - 1) / dd); };

  // ---- one-time packs / inits ----
  pack_w_kernel<<<cdiv(1L * 1024 * 32, 256), 256, 0, stream>>>(encWih_f, p_eih_f, 1024, PTD, 1024, 1);
  pack_w_kernel<<<cdiv(1L * 1024 * 32, 256), 256, 0, stream>>>(encWih_b, p_eih_b, 1024, PTD, 1024, 1);
  pack_w_kernel<<<cdiv(8L * 1024 * 32, 256), 256, 0, stream>>>(encWhh_f, p_ehh_f, 1024, EHD, 1024, 8);
  pack_w_kernel<<<cdiv(8L * 1024 * 32, 256), 256, 0, stream>>>(encWhh_b, p_ehh_b, 1024, EHD, 1024, 8);
  pack_w_kernel<<<cdiv(5L * 2048 * 32, 256), 256, 0, stream>>>(decWih, p_dih, 2048, PTD + ZSD, 2048, 5);
  pack_w_kernel<<<cdiv(16L * 2048 * 32, 256), 256, 0, stream>>>(decWhh, p_dhh, 2048, DHD, 2048, 16);
  pack_w_kernel<<<cdiv(16L * 128 * 32, 256), 256, 0, stream>>>(Wdec, p_wdec, 6 * NMIX + 3, DHD, 128, 16);
  pack_xenc_kernel<<<cdiv((long)SEQL * BATCH * 32, 256), 256, 0, stream>>>(s, xpad_enc);
  zero_f16_kernel<<<cdiv((long)BATCH * EHD, 256), 256, 0, stream>>>(henc_f0, BATCH * EHD);
  zero_f16_kernel<<<cdiv((long)BATCH * EHD, 256), 256, 0, stream>>>(henc_b0, BATCH * EHD);
  zero_f32_kernel<<<cdiv(2L * BATCH * EHD, 256), 256, 0, stream>>>(cenc, 2 * BATCH * EHD);
  zero_f32_kernel<<<cdiv((long)BATCH * DHD, 256), 256, 0, stream>>>(cdec, BATCH * DHD);
  init_xdec_kernel<<<cdiv((long)BATCH * 160, 256), 256, 0, stream>>>(xpad_dec);

  // ---- encoder: 128 steps, both directions fused per launch, ping-pong h ----
  for (int t = 0; t < SEQL; ++t) {
    const _Float16* hinf = (t & 1) ? henc_f1 : henc_f0;
    _Float16*      houtf = (t & 1) ? henc_f0 : henc_f1;
    const _Float16* hinb = (t & 1) ? henc_b1 : henc_b0;
    _Float16*      houtb = (t & 1) ? henc_b0 : henc_b1;
    enc_step_kernel<<<dim3(64, 2), dim3(128), 0, stream>>>(
        xpad_enc, p_eih_f, p_ehh_f, encb_f, p_eih_b, p_ehh_b, encb_b,
        hinf, houtf, cenc_f, hinb, houtb, cenc_b, t);
  }
  // after 128 steps, final h is in buffer 0 of each direction
  latent_kernel<<<cdiv((long)BATCH * ZSD, 256), 256, 0, stream>>>(
      henc_f0, henc_b0, Wmu, bmu, Wsig, bsig, z_eps, zbuf, xpad_dec);
  h0_kernel<<<cdiv((long)BATCH * DHD, 256), 256, 0, stream>>>(zbuf, Wh0, bh0, hdec0);

  // ---- decoder: 256 steps, gate+cell then head+sampling ----
  for (int t = 0; t < NMAXT; ++t) {
    const _Float16* hin = (t & 1) ? hdec1 : hdec0;
    _Float16*      hout = (t & 1) ? hdec0 : hdec1;
    dec_step_kernel<<<dim3(128), dim3(128), 0, stream>>>(
        xpad_dec, p_dih, p_dhh, decb, hin, hout, cdec);
    dec_out_kernel<<<dim3(16), dim3(256), 0, stream>>>(
        hout, p_wdec, bdec, gmm_eps, pen_u, out, xpad_dec, t);
  }
}
